// CuPyLinear_3246995276086
// MI455X (gfx1250) — matrix operand order repaired and checked by reading, not compiled, not run on hardware
//
#include <hip/hip_runtime.h>
#include <hip/hip_bf16.h>

typedef __attribute__((ext_vector_type(16))) __bf16 v16bf;
typedef __attribute__((ext_vector_type(8)))  float  v8f;
typedef __attribute__((ext_vector_type(4)))  unsigned int u32x4;
typedef __attribute__((ext_vector_type(8)))  int          i32x8;
typedef __attribute__((ext_vector_type(4)))  int          i32x4;

#define IN_F   4096
#define OUT_F  4096
#define M_TOT  4096   /* BATCH*SEQ = 2*2048 */

#define BM 128
#define BN 128
#define BK 32
#define LDS_STRIDE 40  /* elements: 64B row + 16B TDM pad = 80B, bank-conflict-free */
#define TILE_BYTES (BM * LDS_STRIDE * 2)   /* 10240 B per staged tile */

// ---------------- kernel 1: scatter COO -> dense f32 W ----------------
__global__ void scatter_coo(const float* __restrict__ data,
                            const int* __restrict__ rows,
                            const int* __restrict__ cols,
                            float* __restrict__ W, int nnz) {
  int i = blockIdx.x * blockDim.x + threadIdx.x;
  int stride = gridDim.x * blockDim.x;
  for (; i < nnz; i += stride) {
    unsafeAtomicAdd(&W[(size_t)rows[i] * IN_F + cols[i]], data[i]);
  }
}

// ---------------- kernel 2: f32 -> bf16 (8 elems / thread, b128 I/O) ----------------
__global__ void cvt_f32_bf16(const float* __restrict__ src,
                             __hip_bfloat16* __restrict__ dst, int n) {
  int i = (blockIdx.x * blockDim.x + threadIdx.x) * 8;
  int stride = gridDim.x * blockDim.x * 8;
  for (; i < n; i += stride) {
    float4 f0 = ((const float4*)(src + i))[0];
    float4 f1 = ((const float4*)(src + i))[1];
    union { __hip_bfloat16 h[8]; uint4 q; } o;
    o.h[0] = __float2bfloat16(f0.x); o.h[1] = __float2bfloat16(f0.y);
    o.h[2] = __float2bfloat16(f0.z); o.h[3] = __float2bfloat16(f0.w);
    o.h[4] = __float2bfloat16(f1.x); o.h[5] = __float2bfloat16(f1.y);
    o.h[6] = __float2bfloat16(f1.z); o.h[7] = __float2bfloat16(f1.w);
    *(uint4*)(dst + i) = o.q;
  }
}

// ---------------- TDM: DMA one 128x32 bf16 tile (row stride 4096) global -> LDS ----------------
// D# per CDNA5 ISA. LDS rows padded by TDM: pad_interval=3 (16 DWORDs
// = one 64B row), pad_amount=3 (4 DWORDs = 16B) -> 80B LDS row stride == LDS_STRIDE.
__device__ __forceinline__ void tdm_load_tile(unsigned lds_byte_off, const void* gptr) {
  unsigned long long ga = (unsigned long long)gptr;
  u32x4 g0;
  g0.x = 1u;                                               // count=1, is_restore=0, no gather
  g0.y = lds_byte_off;                                     // lds_addr [63:32]
  g0.z = (unsigned)(ga & 0xffffffffu);                     // global_addr [95:64]
  g0.w = (unsigned)((ga >> 32) & 0x01ffffffu)              // global_addr [120:96]
         | 0x80000000u;                                    // type=2 ("image") [127:126]
  i32x8 g1;
  g1[0] = (1 << 16)      /* data_size = 2B */
        | (1 << 20)      /* pad_enable */
        | (3 << 22)      /* pad_interval: 16 DWORDs */
        | (3 << 25);     /* pad_amount: 4 DWORDs */
  g1[1] = (int)((IN_F & 0xffff) << 16);                    // tensor_dim0[15:0] @ bits 63:48
  g1[2] = (int)((IN_F >> 16) | ((OUT_F & 0xffff) << 16));  // dim0[31:16] | tensor_dim1[15:0]
  g1[3] = (int)((OUT_F >> 16) | (BK << 16));               // dim1[31:16] | tile_dim0 = 32
  g1[4] = BM;                                              // tile_dim1 = 128, tile_dim2 = 0
  g1[5] = IN_F;                                            // tensor_dim0_stride[31:0]
  g1[6] = 0;                                               // stride[47:32] | dim1_stride lo
  g1[7] = 0;
  i32x4 gz = {0, 0, 0, 0};                                 // groups 2/3 unused (2-D tile)
#if __has_include(<hip/amd_detail/amd_gfx1250_TDM.h>)
  i32x8 gz8 = {0, 0, 0, 0, 0, 0, 0, 0};
  __builtin_amdgcn_tensor_load_to_lds(g0, g1, gz, gz, gz8, 0);   // 6-arg toolchain
#else
  __builtin_amdgcn_tensor_load_to_lds(g0, g1, gz, gz, 0);        // 5-arg toolchain (ROCm 7.2)
#endif
}

// ---------------- kernel 3: Y[M,N] = Xh[M,K] * Wh[N,K]^T via bf16 WMMA + TDM pipeline ----------------
union FragAB { v16bf v; uint2 q[4]; };

__global__ __launch_bounds__(256)
void gemm_bf16_wmma(const __hip_bfloat16* __restrict__ X,   // [M_TOT][IN_F] bf16
                    const __hip_bfloat16* __restrict__ Wt,  // [OUT_F][IN_F] bf16
                    float* __restrict__ Y) {                // [M_TOT][OUT_F] f32
  extern __shared__ unsigned short smem[];  // [A0 | A1 | B0 | B1], dynamic-only => offset 0

  const int tid  = threadIdx.x;
  const int lane = tid & 31;
  const int wave = tid >> 5;
  const int wm   = wave & 3;   // 0..3 -> 32-row strip
  const int wn   = wave >> 2;  // 0..1 -> 64-col strip
  const int m0   = blockIdx.y * BM;
  const int n0   = blockIdx.x * BN;
  const int half = lane >> 4;
  const int l16  = lane & 15;

  const unsigned short* Xu = (const unsigned short*)X;
  const unsigned short* Wu = (const unsigned short*)Wt;

  v8f acc[2][4];
  {
    const v8f vzero = {};
#pragma unroll
    for (int i = 0; i < 2; ++i)
#pragma unroll
      for (int j = 0; j < 4; ++j) acc[i][j] = vzero;
  }

  const int T = IN_F / BK;  // 128 K-tiles

  // prologue: stage tile 0 into buffer 0 (wave0 -> A, wave1 -> B; TDM ignores EXEC,
  // exactly one wave must issue each descriptor)
  if (wave == 0) {
    tdm_load_tile(0u, Xu + (size_t)m0 * IN_F);
    __builtin_amdgcn_s_wait_tensorcnt(0);
  } else if (wave == 1) {
    tdm_load_tile(2u * TILE_BYTES, Wu + (size_t)n0 * IN_F);
    __builtin_amdgcn_s_wait_tensorcnt(0);
  }
  __syncthreads();

  for (int t = 0; t < T; ++t) {
    const int buf = t & 1;
    // issue next tile's DMA into the other buffer; its last readers finished
    // before the barrier that ended iteration t-1, so no WAR hazard
    if (t + 1 < T) {
      const int nbuf = buf ^ 1;
      if (wave == 0)
        tdm_load_tile((unsigned)(nbuf * TILE_BYTES),
                      Xu + (size_t)m0 * IN_F + (t + 1) * BK);
      else if (wave == 1)
        tdm_load_tile((unsigned)(2 * TILE_BYTES + nbuf * TILE_BYTES),
                      Wu + (size_t)n0 * IN_F + (t + 1) * BK);
    }

    const unsigned short* As = smem + buf * (TILE_BYTES / 2);
    const unsigned short* Bs = smem + (2 + buf) * (TILE_BYTES / 2);

    // ---- fragments in contiguous k order (k = 16*half + s), same for A and B ----
    FragAB a[2];
#pragma unroll
    for (int i = 0; i < 2; ++i) {
      const unsigned short* base = As + (wm * 32 + i * 16 + l16) * LDS_STRIDE;
      a[i].q[0] = *(const uint2*)(base + half * 16);
      a[i].q[1] = *(const uint2*)(base + half * 16 + 4);
      a[i].q[2] = *(const uint2*)(base + half * 16 + 8);
      a[i].q[3] = *(const uint2*)(base + half * 16 + 12);
    }
    FragAB b[4];
#pragma unroll
    for (int j = 0; j < 4; ++j) {
      const unsigned short* base = Bs + (wn * 64 + j * 16 + l16) * LDS_STRIDE + half * 16;
      b[j].q[0] = *(const uint2*)(base);
      b[j].q[1] = *(const uint2*)(base + 4);
      b[j].q[2] = *(const uint2*)(base + 8);
      b[j].q[3] = *(const uint2*)(base + 12);
    }

    // ---- 8 WMMAs per wave per K-tile (overlapped with next tile's TDM) ----
#pragma unroll
    for (int i = 0; i < 2; ++i)
#pragma unroll
      for (int j = 0; j < 4; ++j)
        acc[i][j] = __builtin_amdgcn_wmma_f32_16x16x32_bf16(
            false, a[i].v, false, b[j].v, (short)0, acc[i][j], false, false);

    // issuing waves wait for tile t+1's DMA, then everyone syncs (one barrier/tile)
    if (wave < 2) __builtin_amdgcn_s_wait_tensorcnt(0);
    __syncthreads();
  }

  // ---- epilogue: C/D layout VGPR r, lane L -> M = 8*(L>=16)+r, N = L%16 ----
#pragma unroll
  for (int i = 0; i < 2; ++i)
#pragma unroll
    for (int j = 0; j < 4; ++j) {
      const int mb = m0 + wm * 32 + i * 16 + half * 8;
      const int nb = n0 + wn * 64 + j * 16 + l16;
#pragma unroll
      for (int r = 0; r < 8; ++r)
        Y[(size_t)(mb + r) * OUT_F + nb] = acc[i][j][r];
    }
}

extern "C" void kernel_launch(void* const* d_in, const int* in_sizes, int n_in,
                              void* d_out, int out_size, void* d_ws, size_t ws_size,
                              hipStream_t stream) {
  const float* x    = (const float*)d_in[0];   // [2,2048,4096] f32
  const float* data = (const float*)d_in[1];   // [NNZ] f32
  const int*   rows = (const int*)d_in[2];     // [NNZ] i32
  const int*   cols = (const int*)d_in[3];     // [NNZ] i32
  float*       Y    = (float*)d_out;           // [2,2048,4096] f32

  char* ws = (char*)d_ws;
  float*          Wf = (float*)ws;                                               // 64 MB
  __hip_bfloat16* Wh = (__hip_bfloat16*)(ws + (size_t)OUT_F * IN_F * 4);         // 32 MB
  __hip_bfloat16* Xh = (__hip_bfloat16*)(ws + (size_t)OUT_F * IN_F * 4
                                            + (size_t)OUT_F * IN_F * 2);         // 32 MB

  hipMemsetAsync(Wf, 0, (size_t)OUT_F * IN_F * sizeof(float), stream);

  const int nnz = in_sizes[1];
  scatter_coo<<<(nnz + 255) / 256, 256, 0, stream>>>(data, rows, cols, Wf, nnz);

  const int nW = OUT_F * IN_F;
  cvt_f32_bf16<<<(nW / 8 + 255) / 256, 256, 0, stream>>>(Wf, Wh, nW);
  const int nX = M_TOT * IN_F;
  cvt_f32_bf16<<<(nX / 8 + 255) / 256, 256, 0, stream>>>(x, Xh, nX);

  dim3 grid(OUT_F / BN, M_TOT / BM);
  const size_t lds_bytes = 4 * TILE_BYTES;   // double-buffered A + B tiles (40 KB)
  gemm_bf16_wmma<<<grid, 256, lds_bytes, stream>>>(Xh, Wh, Y);
}
